// ZeroMergeAttention_69552700392183
// MI455X (gfx1250) — matrix operand order repaired and checked by reading, not compiled, not run on hardware
//
#include <hip/hip_runtime.h>
#include <hip/hip_bf16.h>

#define SEQ   2048
#define EMBED 1024
#define HEADS 16
#define DH    64
#define B_C   28
#define B_R   14
#define NKEYS (B_C + B_R + 1)   // 43
#define KPITCH 65               // 64 + 1 pad -> conflict-free per-lane rows
#define NEGV  (-1e30f)
#define DECAYF 0.98f
#define ALPHAF 0.6f

typedef __attribute__((ext_vector_type(16))) _Float16 v16h;
typedef __attribute__((ext_vector_type(8)))  float    v8f;

// ---------------------------------------------------------------------------
// WMMA f16 fragment loaders.
// A-matrix 16x32 (MxK), k order used here (contiguous, not ISA table): lane m (0-15): K=[0..15];
// lane m+16: K=[16..31].  (f32 memory, converted in flight;
// amortized over 4 WMMAs.)
// B-matrix 32x16 (KxN), same k order as A: lane n (0-15): K=[0..15];
// lane n+16: K=[16..31].  (pre-converted f16 memory, one aligned 32B load.)
// ---------------------------------------------------------------------------
__device__ __forceinline__ v16h load_a_f32(const float* __restrict__ rowp,
                                           int k0, int lane) {
  const int kb = k0 + ((lane & 16) ? 16 : 0);
  v16h r;
#pragma unroll
  for (int i = 0; i < 8; ++i) r[i] = (_Float16)rowp[kb + i];
#pragma unroll
  for (int i = 0; i < 8; ++i) r[8 + i] = (_Float16)rowp[kb + 8 + i];
  return r;
}

__device__ __forceinline__ v16h load_b_f16(const _Float16* __restrict__ colp,
                                           int k0, int lane) {
  const int kb = k0 + ((lane & 16) ? 16 : 0);
  return *(const v16h*)(colp + kb);   // 32B aligned: kb % 16 == 0, rows 2KB
}

// Wave computes a 16(M) x 64(N) strip: 4 accumulators share one A fragment,
// 4 v_wmma per A load; B fragments come straight from f16 memory (no cvt).
__device__ __forceinline__ void gemm_wave_16x64(
    const float* __restrict__ arow, const _Float16* __restrict__ whbase,
    int n0, int lane, v8f acc[4]) {
  const _Float16* brow = whbase + (size_t)(n0 + (lane & 15)) * EMBED;
  for (int k0 = 0; k0 < EMBED; k0 += 32) {
    v16h a = load_a_f32(arow, k0, lane);
#pragma unroll
    for (int tt = 0; tt < 4; ++tt) {
      v16h b = load_b_f16(brow + (size_t)tt * 16 * EMBED, k0, lane);
      acc[tt] = __builtin_amdgcn_wmma_f32_16x16x32_f16(
          false, a, false, b, (short)0, acc[tt], false, false);
    }
  }
}

// ---------------------------------------------------------------------------
// Kernel 0: pack the four weight matrices to f16 once (bandwidth-trivial).
// wh = [wq | wk | wv | wo], each 1024x1024 halves.
// ---------------------------------------------------------------------------
__global__ __launch_bounds__(256) void cvt_w_kernel(
    const float* __restrict__ wq, const float* __restrict__ wk,
    const float* __restrict__ wv, const float* __restrict__ wo,
    _Float16* __restrict__ wh) {
  const size_t idx = ((size_t)blockIdx.x * 256 + threadIdx.x) * 8;
  const size_t mat = idx >> 20;                 // 1M elements per matrix
  const size_t off = idx & ((1u << 20) - 1);
  const float* s;
  if (mat == 0)      s = wq + off;
  else if (mat == 1) s = wk + off;
  else if (mat == 2) s = wv + off;
  else               s = wo + off;
  _Float16* d = wh + idx;
#pragma unroll
  for (int i = 0; i < 8; ++i) d[i] = (_Float16)s[i];
}

// ---------------------------------------------------------------------------
// Kernel 1: fused Q/K/V projection.  y = x @ W.T + b
// grid = (SEQ/16, EMBED/512, 3), block = 256 (8 waves; wave owns a 64-col
// strip == exactly one head).  Output written as [head][t][d] (f32).
// ---------------------------------------------------------------------------
__global__ __launch_bounds__(256) void qkv_proj_kernel(
    const float* __restrict__ x, const _Float16* __restrict__ wh,
    const float* __restrict__ bq, const float* __restrict__ bk,
    const float* __restrict__ bv,
    float* __restrict__ qh, float* __restrict__ kh, float* __restrict__ vh) {
  const int lane = threadIdx.x & 31;
  const int wid  = threadIdx.x >> 5;
  const int m0 = blockIdx.x * 16;
  const int n0 = (blockIdx.y * 8 + wid) * 64;   // 64-col strip, head-aligned

  const _Float16* w; const float* bias; float* o;
  if (blockIdx.z == 0)      { w = wh;                              bias = bq; o = qh; }
  else if (blockIdx.z == 1) { w = wh + (size_t)EMBED * EMBED;      bias = bk; o = kh; }
  else                      { w = wh + (size_t)2 * EMBED * EMBED;  bias = bv; o = vh; }

  const float* arow = x + (size_t)(m0 + (lane & 15)) * EMBED;

  v8f acc[4] = {};
  gemm_wave_16x64(arow, w, n0, lane, acc);

  const int h = n0 >> 6;  // whole strip lies in one head
  float* ohead = o + (size_t)h * SEQ * DH;
  const int mbase = m0 + ((lane & 16) ? 8 : 0);
#pragma unroll
  for (int tt = 0; tt < 4; ++tt) {
    const int col = n0 + tt * 16 + (lane & 15);
    const int d = col & 63;
    const float bv_ = bias[col];
#pragma unroll
    for (int i = 0; i < 8; ++i) {
      ohead[(size_t)(mbase + i) * DH + d] = acc[tt][i] + bv_;
    }
  }
}

// ---------------------------------------------------------------------------
// Kernel 2: sequential per-head scan.  One wave32 per head; all carry state
// in LDS.  Lane l owns key slots l and l+32 (43 slots total).
// ---------------------------------------------------------------------------
__global__ __launch_bounds__(32) void scan_kernel(
    const float* __restrict__ qh, const float* __restrict__ kh,
    const float* __restrict__ vh, float* __restrict__ attn /*[SEQ][EMBED]*/) {
  __shared__ float keys[NKEYS][KPITCH];
  __shared__ float vals[NKEYS][KPITCH];
  __shared__ float cs[B_C];
  __shared__ float rc[B_R];
  __shared__ float qb[DH], kb[DH], vb[DH];
  __shared__ float evk[DH], evv[DH];
  __shared__ float wb[NKEYS];

  const int h = blockIdx.x;
  const int l = threadIdx.x;
  const float scale = 0.125f;  // 1/sqrt(DH)

  const float* q = qh + (size_t)h * SEQ * DH;
  const float* k = kh + (size_t)h * SEQ * DH;
  const float* v = vh + (size_t)h * SEQ * DH;

  // ---- init carry (t = 0) ----
  for (int i = l; i < NKEYS * KPITCH; i += 32) {
    (&keys[0][0])[i] = 0.f;
    (&vals[0][0])[i] = 0.f;
  }
  __syncthreads();
  for (int d = l; d < DH; d += 32) {
    const float k0v = k[d], v0v = v[d];
    keys[0][d] = k0v;         vals[0][d] = v0v;         // cache slot 0
    keys[NKEYS - 1][d] = k0v; vals[NKEYS - 1][d] = v0v; // pk / pv
    attn[(size_t)h * DH + d] = 0.f;                     // output row 0 is zero
  }
  if (l < B_C) cs[l] = (l == 0) ? 1.0f : NEGV;
  if (l < B_R) rc[l] = 0.f;
  __syncthreads();

  const int j0 = l, j1 = l + 32;

  for (int t = 1; t < SEQ; ++t) {
    for (int d = l; d < DH; d += 32) {
      qb[d] = q[(size_t)t * DH + d];
      kb[d] = k[(size_t)t * DH + d];
      vb[d] = v[(size_t)t * DH + d];
    }
    if (t + 1 < SEQ) {
      __builtin_prefetch(&q[(size_t)(t + 1) * DH], 0, 0);
      __builtin_prefetch(&k[(size_t)(t + 1) * DH], 0, 0);
      __builtin_prefetch(&v[(size_t)(t + 1) * DH], 0, 0);
    }
    __syncthreads();

    // ---- pass 1: masked softmax over old {cache, residual, pk} ----
    float lg0, lg1 = NEGV;
    {
      float acc = 0.f;
      for (int d = 0; d < DH; ++d) acc += keys[j0][d] * qb[d];
      const bool m = (j0 < B_C) ? (cs[j0] > NEGV * 0.5f)
                                : ((j0 < B_C + B_R) ? (rc[j0 - B_C] > 0.f) : true);
      lg0 = m ? acc * scale : NEGV;
    }
    if (j1 < NKEYS) {
      float acc = 0.f;
      for (int d = 0; d < DH; ++d) acc += keys[j1][d] * qb[d];
      const bool m = (j1 < B_C + B_R) ? (rc[j1 - B_C] > 0.f) : true;
      lg1 = m ? acc * scale : NEGV;
    }
    float mx = fmaxf(lg0, lg1);
    for (int o = 16; o > 0; o >>= 1) mx = fmaxf(mx, __shfl_xor(mx, o));
    const float e0 = __expf(lg0 - mx);
    const float e1 = (j1 < NKEYS) ? __expf(lg1 - mx) : 0.f;
    float sm = e0 + e1;
    for (int o = 16; o > 0; o >>= 1) sm += __shfl_xor(sm, o);
    const float w0 = e0 / sm;

    // ---- decay + accumulate attention mass into cache scores ----
    float myc = NEGV;
    if (l < B_C) {
      const float c = cs[l];
      myc = (c > NEGV * 0.5f) ? (c * DECAYF + w0) : NEGV;
      cs[l] = myc;
    }

    // ---- eviction: min over {cs[0..27], 1.0(new)}; ties evict higher idx ----
    float cvv = (l < B_C) ? myc : ((l == B_C) ? 1.0f : 3.4e38f);
    int   ci  = l;
    for (int o = 16; o > 0; o >>= 1) {
      const float ov = __shfl_xor(cvv, o);
      const int   oi = __shfl_xor(ci, o);
      if (ov < cvv || (ov == cvv && oi > ci)) { cvv = ov; ci = oi; }
    }
    const int   ev       = ci;
    const float ev_s     = cvv;
    const bool  ev_valid = ev_s > NEGV * 0.5f;

    __syncthreads();
    for (int d = l; d < DH; d += 32) {
      evk[d] = (ev < B_C) ? keys[ev][d] : kb[d];
      evv[d] = (ev < B_C) ? vals[ev][d] : vb[d];
    }
    __syncthreads();
    if (ev < B_C) {  // insert new token into the freed cache slot
      for (int d = l; d < DH; d += 32) { keys[ev][d] = kb[d]; vals[ev][d] = vb[d]; }
      if (l == 0) cs[ev] = 1.0f;
    }
    __syncthreads();

    // ---- merge evicted key into residual slots (lane r handles slot r) ----
    float simv = NEGV;
    int   femy = 64;
    if (l < B_R) {
      float dot = 0.f, n2 = 0.f, e2 = 0.f;
      for (int d = 0; d < DH; ++d) {
        const float rkd = keys[B_C + l][d], ekd = evk[d];
        dot += rkd * ekd; n2 += rkd * rkd; e2 += ekd * ekd;
      }
      const bool empty = rc[l] <= 0.f;
      simv = empty ? NEGV : dot / (__builtin_sqrtf(e2) * __builtin_sqrtf(n2) + 1e-8f);
      if (empty) femy = l;
    }
    float bvv = simv; int bi = l;
    for (int o = 16; o > 0; o >>= 1) {
      const float ov = __shfl_xor(bvv, o);
      const int   oi = __shfl_xor(bi, o);
      if (ov > bvv || (ov == bvv && oi < bi)) { bvv = ov; bi = oi; }
    }
    int fe = femy;
    for (int o = 16; o > 0; o >>= 1) fe = min(fe, __shfl_xor(fe, o));
    const int   ridx = (fe < 64) ? fe : bi;
    const float cnt  = rc[ridx];
    if (ev_valid) {
      const float invc = 1.f / (cnt + 1.f);
      for (int d = l; d < DH; d += 32) {
        keys[B_C + ridx][d] = (cnt * keys[B_C + ridx][d] + evk[d]) * invc;
        vals[B_C + ridx][d] = (cnt * vals[B_C + ridx][d] + evv[d]) * invc;
      }
      if (l == 0) rc[ridx] = cnt + 1.f;
    }
    // pk, pv <- current token
    for (int d = l; d < DH; d += 32) { keys[NKEYS - 1][d] = kb[d]; vals[NKEYS - 1][d] = vb[d]; }
    __syncthreads();

    // ---- pass 2: count-biased masked softmax, weighted value sum ----
    float lg0b, lg1b = NEGV;
    {
      float acc = 0.f;
      for (int d = 0; d < DH; ++d) acc += keys[j0][d] * qb[d];
      bool m; float cn;
      if (j0 < B_C)            { m = cs[j0] > NEGV * 0.5f; cn = 1.f; }
      else if (j0 < B_C + B_R) { const float r = rc[j0 - B_C]; m = r > 0.f; cn = fmaxf(r, 1.f); }
      else                     { m = true; cn = 1.f; }
      lg0b = m ? (acc * scale + ALPHAF * __logf(cn)) : NEGV;
    }
    if (j1 < NKEYS) {
      float acc = 0.f;
      for (int d = 0; d < DH; ++d) acc += keys[j1][d] * qb[d];
      bool m; float cn;
      if (j1 < B_C + B_R) { const float r = rc[j1 - B_C]; m = r > 0.f; cn = fmaxf(r, 1.f); }
      else                { m = true; cn = 1.f; }
      lg1b = m ? (acc * scale + ALPHAF * __logf(cn)) : NEGV;
    }
    float mx2 = fmaxf(lg0b, lg1b);
    for (int o = 16; o > 0; o >>= 1) mx2 = fmaxf(mx2, __shfl_xor(mx2, o));
    const float e0b = __expf(lg0b - mx2);
    const float e1b = (j1 < NKEYS) ? __expf(lg1b - mx2) : 0.f;
    float s2 = e0b + e1b;
    for (int o = 16; o > 0; o >>= 1) s2 += __shfl_xor(s2, o);
    const float inv2 = 1.f / s2;
    wb[j0] = e0b * inv2;
    if (j1 < NKEYS) wb[j1] = e1b * inv2;
    __syncthreads();

    float o0 = 0.f, o1 = 0.f;
    for (int j = 0; j < NKEYS; ++j) {
      const float wj = wb[j];
      o0 += wj * vals[j][2 * l];
      o1 += wj * vals[j][2 * l + 1];
    }
    attn[(size_t)t * EMBED + h * DH + 2 * l]     = o0;
    attn[(size_t)t * EMBED + h * DH + 2 * l + 1] = o1;
    __syncthreads();
  }
}

// ---------------------------------------------------------------------------
// Kernel 3: output projection.  y = attn @ wo.T + bo  ->  d_out [SEQ][EMBED]
// grid = (SEQ/16, EMBED/512), block = 256.
// ---------------------------------------------------------------------------
__global__ __launch_bounds__(256) void out_proj_kernel(
    const float* __restrict__ attn, const _Float16* __restrict__ woh,
    const float* __restrict__ bo, float* __restrict__ out) {
  const int lane = threadIdx.x & 31;
  const int wid  = threadIdx.x >> 5;
  const int m0 = blockIdx.x * 16;
  const int n0 = (blockIdx.y * 8 + wid) * 64;

  const float* arow = attn + (size_t)(m0 + (lane & 15)) * EMBED;

  v8f acc[4] = {};
  gemm_wave_16x64(arow, woh, n0, lane, acc);

  const int mbase = m0 + ((lane & 16) ? 8 : 0);
#pragma unroll
  for (int tt = 0; tt < 4; ++tt) {
    const int col = n0 + tt * 16 + (lane & 15);
    const float bv_ = bo[col];
#pragma unroll
    for (int i = 0; i < 8; ++i) {
      out[(size_t)(mbase + i) * EMBED + col] = acc[tt][i] + bv_;
    }
  }
}

// ---------------------------------------------------------------------------
extern "C" void kernel_launch(void* const* d_in, const int* in_sizes, int n_in,
                              void* d_out, int out_size, void* d_ws, size_t ws_size,
                              hipStream_t stream) {
  (void)in_sizes; (void)n_in; (void)out_size; (void)ws_size;
  const float* x  = (const float*)d_in[0];
  const float* wq = (const float*)d_in[1];
  const float* bq = (const float*)d_in[2];
  const float* wk = (const float*)d_in[3];
  const float* bk = (const float*)d_in[4];
  const float* wv = (const float*)d_in[5];
  const float* bv = (const float*)d_in[6];
  const float* wo = (const float*)d_in[7];
  const float* bo = (const float*)d_in[8];
  float* out = (float*)d_out;

  // workspace layout (40 MB):
  //   q,k,v  f32 [HEADS][SEQ][DH]   (3 x 8 MB)
  //   attn   f32 [SEQ][EMBED]       (8 MB)
  //   wh     f16 [4][EMBED][EMBED]  (8 MB)
  float* qh   = (float*)d_ws;
  float* kh   = qh + (size_t)SEQ * EMBED;
  float* vh   = kh + (size_t)SEQ * EMBED;
  float* attn = vh + (size_t)SEQ * EMBED;
  _Float16* wh = (_Float16*)(attn + (size_t)SEQ * EMBED);

  cvt_w_kernel<<<dim3(4 * EMBED * EMBED / (256 * 8)), 256, 0, stream>>>(
      wq, wk, wv, wo, wh);
  qkv_proj_kernel<<<dim3(SEQ / 16, EMBED / 512, 3), 256, 0, stream>>>(
      x, wh, bq, bk, bv, qh, kh, vh);
  scan_kernel<<<dim3(HEADS), 32, 0, stream>>>(qh, kh, vh, attn);
  out_proj_kernel<<<dim3(SEQ / 16, EMBED / 512), 256, 0, stream>>>(
      attn, wh + (size_t)3 * EMBED * EMBED, bo, out);
}
